// Critic_18056042512980
// MI455X (gfx1250) — matrix operand order repaired and checked by reading, not compiled, not run on hardware
//
#include <hip/hip_runtime.h>

typedef __attribute__((ext_vector_type(16))) _Float16 v16h;
typedef __attribute__((ext_vector_type(4)))  _Float16 v4h;
typedef __attribute__((ext_vector_type(8)))  float    v8f;

// ---------------------------------------------------------------------------
// Packed fragment layouts (wave32, 16x16x32 f16 WMMA), per CDNA5 ISA
//
// A fragment (M=16,K=32), lane = hi*16 + m:
//   lanes 0-15  (m=0..15) hold K = {0..7} U {16..23}   (halves 0..15)
//   lanes 16-31 (m=0..15) hold K = {8..15} U {24..31}
// B fragment (K=32,N=16), lane = hi*16 + n:
//   lanes 0-15 hold K = {0..7} U {16..23}; lanes 16-31 hold K = {8..15} U {24..31} (same as A)
//
// Packed storage: fragment-major [tile][lane][16 halves] so each lane reads
// one contiguous 32B v16h (coalesced 1KB per wave per fragment).
// ---------------------------------------------------------------------------

// flat index of element (row, k) inside packed-A (K fixed at 128 -> 4 k-tiles)
__device__ __forceinline__ size_t a_pack_idx(int row, int k) {
  int tr = row >> 4, m = row & 15;
  int tk = k >> 5,  kk = k & 31;
  int hi = (kk >> 3) & 1;
  int h  = (kk & 7) + ((kk & 16) ? 8 : 0);
  return ((size_t)(tr * 4 + tk) * 32 + (hi * 16 + m)) * 16 + h;
}

__device__ __forceinline__ v16h load_a_packed(const _Float16* __restrict__ Ap,
                                              int row0, int k0, int lane) {
  size_t idx = ((size_t)((row0 >> 4) * 4 + (k0 >> 5)) * 32 + lane) * 16;
  return *(const v16h*)(Ap + idx);
}

// nTN = number of 16-wide column tiles in B
__device__ __forceinline__ v16h load_b_packed(const _Float16* __restrict__ Bp, int nTN,
                                              int k0, int col0, int lane) {
  size_t idx = ((size_t)((k0 >> 5) * nTN + (col0 >> 4)) * 32 + lane) * 16;
  return *(const v16h*)(Bp + idx);
}

// ---------------------------------------------------------------------------
// Utility kernels
// ---------------------------------------------------------------------------
__global__ void zero_f32(float* __restrict__ a, int n) {
  int i = blockIdx.x * blockDim.x + threadIdx.x;
  if (i < n) a[i] = 0.f;
}

__global__ void init_out(float* __restrict__ out) { out[0] = 0.f; }

// f32 weight [K][Ncols] -> f16 packed-B fragments
__global__ void pack_b_f16(const float* __restrict__ W, _Float16* __restrict__ Bp,
                           int K, int Ncols) {
  int i = blockIdx.x * blockDim.x + threadIdx.x;
  if (i >= K * Ncols) return;
  int k = i / Ncols, n = i % Ncols;
  int tk = k >> 5, kk = k & 31;
  int tn = n >> 4, nn = n & 15;
  int hi = (kk >> 3) & 1, h = (kk & 7) + ((kk & 16) ? 8 : 0);
  int nTN = Ncols >> 4;
  size_t dst = ((size_t)(tk * nTN + tn) * 32 + (hi * 16 + nn)) * 16 + h;
  Bp[dst] = (_Float16)W[i];
}

// out-degree over src (feature), in-degree over dst (GCN norm & SAGE count)
__global__ void degree_kernel(const int* __restrict__ src, const int* __restrict__ dst,
                              float* __restrict__ degS, float* __restrict__ degD, int E) {
  int e = blockIdx.x * blockDim.x + threadIdx.x;
  if (e < E) {
    atomicAdd(&degS[src[e]], 1.f);
    atomicAdd(&degD[dst[e]], 1.f);
  }
}

// g[i] = dinv[i] * (xin[i] @ W_gcn); t initialized with self-loop term g[i]
// one thread = 4 features (float4 path)
__global__ void gcn_node_kernel(const float* __restrict__ x, const float* __restrict__ Wg,
                                const float* __restrict__ degS, const float* __restrict__ degD,
                                float* __restrict__ dinv, float4* __restrict__ g,
                                float4* __restrict__ t, int N) {
  size_t tid = (size_t)blockIdx.x * blockDim.x + threadIdx.x;
  if (tid >= (size_t)N * 32) return;
  int i = (int)(tid >> 5), q = (int)(tid & 31);
  float di = rsqrtf(degD[i] + 1.0f);            // deg incl. self-loop >= 1
  float xi = x[i], d0 = degS[i];
  float4 w0 = *(const float4*)(Wg + q * 4);
  float4 w1 = *(const float4*)(Wg + 128 + q * 4);
  float4 gv;
  gv.x = di * (xi * w0.x + d0 * w1.x);
  gv.y = di * (xi * w0.y + d0 * w1.y);
  gv.z = di * (xi * w0.z + d0 * w1.z);
  gv.w = di * (xi * w0.w + d0 * w1.w);
  g[tid] = gv;
  t[tid] = gv;                                  // self-loop contribution
  if (q == 0) dinv[i] = di;
}

// to[dst] += from[src]: one wave per edge, one float4 per lane.
// e is wave-uniform -> src/dst resolve to scalar loads.
__global__ __launch_bounds__(256) void edge_scatter_kernel(
    const int* __restrict__ src, const int* __restrict__ dst,
    const float4* __restrict__ from, float* __restrict__ to, int E) {
  int e = blockIdx.x * 8 + (threadIdx.x >> 5);
  if (e >= E) return;
  int lane = threadIdx.x & 31;
  int s = src[e], d = dst[e];
  float4 v = from[(size_t)s * 32 + lane];
  float* tp = to + (size_t)d * 128 + lane * 4;
  atomicAdd(tp + 0, v.x);
  atomicAdd(tp + 1, v.y);
  atomicAdd(tp + 2, v.z);
  atomicAdd(tp + 3, v.w);
}

// a1 = relu(dinv*t + b_gcn); t (f32) kept in place for SAGE scatter;
// a1h written in packed-A layout; aggsum zeroed for the SAGE pass.
__global__ void a1_kernel(float4* __restrict__ t, const float* __restrict__ dinv,
                          const float* __restrict__ bg, _Float16* __restrict__ a1h,
                          float4* __restrict__ aggsum, int N) {
  size_t tid = (size_t)blockIdx.x * blockDim.x + threadIdx.x;
  if (tid >= (size_t)N * 32) return;
  int i = (int)(tid >> 5), q = (int)(tid & 31);
  float di = dinv[i];
  float4 tv = t[tid];
  float4 bv = *(const float4*)(bg + q * 4);
  float4 a;
  a.x = fmaxf(di * tv.x + bv.x, 0.f);
  a.y = fmaxf(di * tv.y + bv.y, 0.f);
  a.z = fmaxf(di * tv.z + bv.z, 0.f);
  a.w = fmaxf(di * tv.w + bv.w, 0.f);
  t[tid] = a;                                   // a1 (f32) for SAGE scatter
  v4h h4; h4[0] = (_Float16)a.x; h4[1] = (_Float16)a.y;
  h4[2] = (_Float16)a.z; h4[3] = (_Float16)a.w;
  *(v4h*)(a1h + a_pack_idx(i, q * 4)) = h4;     // contiguous 8B in packed-A
  float4 zz = {0.f, 0.f, 0.f, 0.f};
  aggsum[tid] = zz;                             // reset accumulator (reused buffer)
}

// agg = aggsum / max(in_degree,1) -> packed-A f16
__global__ void agg_kernel(const float4* __restrict__ aggsum, const float* __restrict__ degD,
                           _Float16* __restrict__ aggh, int N) {
  size_t tid = (size_t)blockIdx.x * blockDim.x + threadIdx.x;
  if (tid >= (size_t)N * 32) return;
  int i = (int)(tid >> 5), q = (int)(tid & 31);
  float inv = 1.0f / fmaxf(degD[i], 1.f);
  float4 s = aggsum[tid];
  v4h h4;
  h4[0] = (_Float16)(s.x * inv); h4[1] = (_Float16)(s.y * inv);
  h4[2] = (_Float16)(s.z * inv); h4[3] = (_Float16)(s.w * inv);
  *(v4h*)(aggh + a_pack_idx(i, q * 4)) = h4;
}

// ---------------------------------------------------------------------------
// a2 = agg @ W_sage_l + a1 @ W_sage_r + b_sage_l  (dual WMMA, f32 accum)
// block = 256 = 8 waves; wave w -> cols [16w,16w+16); 16 rows per block.
// a2 written in packed-A layout (f16) for the head GEMM.
// ---------------------------------------------------------------------------
__global__ __launch_bounds__(256) void sage_gemm_kernel(
    const _Float16* __restrict__ aggh, const _Float16* __restrict__ a1h,
    const _Float16* __restrict__ Wlp, const _Float16* __restrict__ Wrp,
    const float* __restrict__ bias, _Float16* __restrict__ a2h, int N) {
  int lane = threadIdx.x & 31;
  int wave = threadIdx.x >> 5;
  int row0 = blockIdx.x << 4;
  int col0 = wave << 4;
  v8f c = {};
#pragma unroll
  for (int k0 = 0; k0 < 128; k0 += 32) {
    v16h aL = load_a_packed(aggh, row0, k0, lane);
    v16h bL = load_b_packed(Wlp, 8, k0, col0, lane);
    c = __builtin_amdgcn_wmma_f32_16x16x32_f16(false, aL, false, bL, (short)0, c, false, false);
    v16h aR = load_a_packed(a1h, row0, k0, lane);
    v16h bR = load_b_packed(Wrp, 8, k0, col0, lane);
    c = __builtin_amdgcn_wmma_f32_16x16x32_f16(false, aR, false, bR, (short)0, c, false, false);
  }
  int n  = col0 + (lane & 15);
  int hi = (lane >> 4) & 1;
  float bn = bias[n];
#pragma unroll
  for (int r = 0; r < 8; ++r) {
    int m = row0 + r + (hi ? 8 : 0);
    if (m < N) a2h[a_pack_idx(m, n)] = (_Float16)(c[r] + bn);
  }
}

// ---------------------------------------------------------------------------
// out = sum_i relu(a2 @ W_v1 + b_v1) @ W_v2 + N*b_v2
// block = 128 = 4 waves covering 64 head cols; 16 rows/block.
// GEMV + global sum fold into per-lane partials -> LDS tree -> 1 atomic/block.
// ---------------------------------------------------------------------------
__global__ __launch_bounds__(128) void head_kernel(
    const _Float16* __restrict__ a2h, const _Float16* __restrict__ Wv1p,
    const float* __restrict__ bv1, const float* __restrict__ Wv2,
    const float* __restrict__ bv2, float* __restrict__ out, int N) {
  __shared__ float red[128];
  int lane = threadIdx.x & 31;
  int wave = threadIdx.x >> 5;      // 0..3
  int row0 = blockIdx.x << 4;
  int col0 = wave << 4;
  v8f c = {};
#pragma unroll
  for (int k0 = 0; k0 < 128; k0 += 32) {
    v16h a = load_a_packed(a2h, row0, k0, lane);
    v16h b = load_b_packed(Wv1p, 4, k0, col0, lane);
    c = __builtin_amdgcn_wmma_f32_16x16x32_f16(false, a, false, b, (short)0, c, false, false);
  }
  int n  = col0 + (lane & 15);
  int hi = (lane >> 4) & 1;
  float w2 = Wv2[n];
  float bn = bv1[n];
  float partial = 0.f;
#pragma unroll
  for (int r = 0; r < 8; ++r) {
    int m = row0 + r + (hi ? 8 : 0);
    if (m < N) {
      float t = c[r] + bn;
      t = t > 0.f ? t : 0.f;
      partial += t * w2;
    }
  }
  red[threadIdx.x] = partial;
  __syncthreads();
  for (int s = 64; s > 0; s >>= 1) {
    if (threadIdx.x < s) red[threadIdx.x] += red[threadIdx.x + s];
    __syncthreads();
  }
  if (threadIdx.x == 0) {
    float add = red[0];
    if (blockIdx.x == 0) add += (float)N * bv2[0];   // + sum_i b_v2
    atomicAdd(out, add);
  }
}

// ---------------------------------------------------------------------------
extern "C" void kernel_launch(void* const* d_in, const int* in_sizes, int n_in,
                              void* d_out, int out_size, void* d_ws, size_t ws_size,
                              hipStream_t stream) {
  const float* x      = (const float*)d_in[0];
  const int*   ei     = (const int*)d_in[1];
  const float* W_gcn  = (const float*)d_in[2];
  const float* b_gcn  = (const float*)d_in[3];
  const float* W_sl   = (const float*)d_in[4];
  const float* b_sl   = (const float*)d_in[5];
  const float* W_sr   = (const float*)d_in[6];
  const float* W_v1   = (const float*)d_in[7];
  const float* b_v1   = (const float*)d_in[8];
  const float* W_v2   = (const float*)d_in[9];
  const float* b_v2   = (const float*)d_in[10];
  float* out = (float*)d_out;

  const int N = in_sizes[0];
  const int E = in_sizes[1] / 2;
  const int* src = ei;
  const int* dst = ei + E;
  const int rowTiles = (N + 15) / 16;
  const size_t Npad = (size_t)rowTiles * 16;

  // ---- workspace carve-out (256B aligned) ----
  char* p = (char*)d_ws;
  auto carve = [&](size_t bytes) -> void* {
    void* r = (void*)p;
    p += (bytes + 255) & ~(size_t)255;
    return r;
  };
  float*    degS  = (float*)carve((size_t)2 * N * sizeof(float)); // degS+degD contiguous
  float*    degD  = degS + N;
  float*    dinv  = (float*)carve((size_t)N * sizeof(float));
  float*    bufA  = (float*)carve((size_t)N * 128 * sizeof(float)); // g, then aggsum
  float*    bufB  = (float*)carve((size_t)N * 128 * sizeof(float)); // t, then a1 (f32)
  _Float16* a1h   = (_Float16*)carve(Npad * 128 * sizeof(_Float16)); // packed-A
  _Float16* aggh  = (_Float16*)carve(Npad * 128 * sizeof(_Float16)); // packed-A
  _Float16* a2h   = (_Float16*)carve(Npad * 128 * sizeof(_Float16)); // packed-A
  _Float16* Wlp   = (_Float16*)carve(128 * 128 * sizeof(_Float16));  // packed-B
  _Float16* Wrp   = (_Float16*)carve(128 * 128 * sizeof(_Float16));  // packed-B
  _Float16* Wv1p  = (_Float16*)carve(128 * 64 * sizeof(_Float16));   // packed-B

  const int B256 = 256;
  const size_t NT = (size_t)N * 32;                 // node x feature/4 threads
  const int gridNT = (int)((NT + B256 - 1) / B256);
  const int gridE8 = (E + 7) / 8;                   // 8 edges (waves) per block

  // 0) init
  zero_f32<<<(2 * N + B256 - 1) / B256, B256, 0, stream>>>(degS, 2 * N);
  init_out<<<1, 1, 0, stream>>>(out);

  // 1) degrees
  degree_kernel<<<(E + B256 - 1) / B256, B256, 0, stream>>>(src, dst, degS, degD, E);

  // 2) weights -> f16 packed-B fragments
  pack_b_f16<<<(128 * 128 + B256 - 1) / B256, B256, 0, stream>>>(W_sl, Wlp, 128, 128);
  pack_b_f16<<<(128 * 128 + B256 - 1) / B256, B256, 0, stream>>>(W_sr, Wrp, 128, 128);
  pack_b_f16<<<(128 * 64 + B256 - 1) / B256, B256, 0, stream>>>(W_v1, Wv1p, 128, 64);

  // 3) GCN node transform: g = dinv * (xin @ W_gcn); t = g (self-loop)
  gcn_node_kernel<<<gridNT, B256, 0, stream>>>(x, W_gcn, degS, degD, dinv,
                                               (float4*)bufA, (float4*)bufB, N);

  // 4) GCN edge scatter: t[dst] += g[src]
  edge_scatter_kernel<<<gridE8, B256, 0, stream>>>(src, dst, (const float4*)bufA, bufB, E);

  // 5) a1 = relu(dinv*t + b_gcn); bufB <- a1; a1h packed; bufA <- 0 (aggsum)
  a1_kernel<<<gridNT, B256, 0, stream>>>((float4*)bufB, dinv, b_gcn, a1h, (float4*)bufA, N);

  // 6) SAGE edge scatter: aggsum[dst] += a1[src]
  edge_scatter_kernel<<<gridE8, B256, 0, stream>>>(src, dst, (const float4*)bufB, bufA, E);

  // 7) agg = aggsum / max(cnt,1) -> packed-A f16
  agg_kernel<<<gridNT, B256, 0, stream>>>((const float4*)bufA, degD, aggh, N);

  // 8) a2 = agg@W_sage_l + a1@W_sage_r + b_sage_l   (WMMA)
  sage_gemm_kernel<<<rowTiles, 256, 0, stream>>>(aggh, a1h, Wlp, Wrp, b_sl, a2h, N);

  // 9) out = sum(relu(a2@W_v1+b_v1)@W_v2) + N*b_v2   (WMMA + reduction)
  head_kernel<<<rowTiles, 128, 0, stream>>>(a2h, Wv1p, b_v1, W_v2, b_v2, out, N);
}
